// modelLSTMcell_22445499089340
// MI455X (gfx1250) — matrix operand order repaired and checked by reading, not compiled, not run on hardware
//
#include <hip/hip_runtime.h>
#include <hip/hip_bf16.h>

// LSTM forward, batch-partitioned persistent kernel for gfx1250 (MI455X).
// NT=365, NGRID=1000, NX=32, NH=256, gates=1024.
// - 63 workgroups x 256 threads (8 wave32s); WG owns 16 batch rows for all 365 steps.
// - Recurrent GEMM h(16x256) @ W_hh^T(256x1024) via V_WMMA_F32_16X16X32_BF16,
//   fp32 accumulate. W_hh/W_ih prepacked once into B-fragment layout in d_ws (576KB).
// - Weight base pointer is laundered per time-step through an empty asm AS AN
//   ADDRESS-SPACE(1) POINTER: blocks LICM (which otherwise hoists 576 VGPRs of
//   fragments and spills to scratch) while keeping GLOBAL (not FLAT) addressing.
// - h kept in LDS (bf16, padded stride) and re-read as A fragments each step.
// - Output projection (NY=1) fused; no hs buffer ever materialized.

#define NTT   365
#define NGRID 1000
#define NXX   32
#define NHH   256

#define GAS __attribute__((address_space(1)))

typedef __attribute__((ext_vector_type(16))) __bf16        v16bf;
typedef __attribute__((ext_vector_type(8)))  float         v8f;
typedef __attribute__((ext_vector_type(4)))  unsigned int  uint4v;  // builtin vector:
// unlike HIP's class-type uint4, assignment works from addrspace-qualified lvalues.

struct BF16Frag { uint4v a, b; };     // 32 bytes == v16bf
struct U16x16   { unsigned short v[16]; };

static __device__ __forceinline__ unsigned short f2bfu(float f) {
  unsigned u = __builtin_bit_cast(unsigned, f);
  unsigned r = u + 0x7FFFu + ((u >> 16) & 1u);   // round-to-nearest-even
  return (unsigned short)(r >> 16);
}
static __device__ __forceinline__ float bf2f(unsigned short s) {
  return __builtin_bit_cast(float, ((unsigned)s) << 16);
}
static __device__ __forceinline__ float sigf(float x) {
  return 1.0f / (1.0f + __expf(-x));
}
// contiguous 32B fragment load from GLOBAL (prepacked weights): lowers to
// global_load_b128 x2 with SGPR base + immediate offset.
static __device__ __forceinline__ v16bf ld_frag_g(const GAS unsigned short* p) {
  BF16Frag r;
  r.a = ((const GAS uint4v*)p)[0];
  r.b = ((const GAS uint4v*)p)[1];
  return __builtin_bit_cast(v16bf, r);
}
// two-part fragment load (LDS A-operand: K runs 0..7 and 16..23 per lane half)
static __device__ __forceinline__ v16bf ld_frag2(const unsigned short* p0,
                                                 const unsigned short* p1) {
  BF16Frag r;
  r.a = *(const uint4v*)p0;
  r.b = *(const uint4v*)p1;
  return __builtin_bit_cast(v16bf, r);
}

// ---------------------------------------------------------------------------
// Prepack W_hh (1024x256) and W_ih (1024x32) fp32 -> bf16 WMMA B-fragments.
// B 32x16 bf16 layout: lane l: N = tile*16 + (l%16); element e maps to
// K = kc*32 + (l<16 ? 0 : 8) + (e<8 ? e : 8+e). Fragment = 32 lanes * 16 bf16 = 1024B.
// ws layout: [512 frags of W_hh][64 frags of W_ih], ushort-addressed.
// ---------------------------------------------------------------------------
__global__ __launch_bounds__(256) void prepack_weights(
    const float* __restrict__ W_ih, const float* __restrict__ W_hh,
    unsigned short* __restrict__ wpack) {
  int tid = blockIdx.x * 256 + threadIdx.x;
  const int NWHH = 64 * 8 * 512;            // 262144 bf16 elems
  const int NWIH = 64 * 512;                // 32768
  if (tid < NWHH) {
    int e = tid & 15, lane = (tid >> 4) & 31, frag = tid >> 9;
    int tile = frag >> 3, kc = frag & 7;
    int K = kc * 32 + ((lane < 16) ? 0 : 8) + ((e < 8) ? e : 8 + e);
    int N = tile * 16 + (lane & 15);
    wpack[frag * 512 + lane * 16 + e] = f2bfu(W_hh[N * NHH + K]);
  } else if (tid < NWHH + NWIH) {
    int t2 = tid - NWHH;
    int e = t2 & 15, lane = (t2 >> 4) & 31, tile = t2 >> 9;
    int K = ((lane < 16) ? 0 : 8) + ((e < 8) ? e : 8 + e);
    int N = tile * 16 + (lane & 15);
    wpack[NWHH + tile * 512 + lane * 16 + e] = f2bfu(W_ih[N * NXX + K]);
  }
}

// ---------------------------------------------------------------------------
// Persistent LSTM kernel. blockIdx.x = batch tile of 16 rows (63 tiles).
// wave w owns hidden units [32w, 32w+32): gate column tiles q*16 + 2w + j,
// q in {i,f,g,o}, j in {0,1}.
// ---------------------------------------------------------------------------
__global__ __launch_bounds__(256) void lstm_persist(
    const float* __restrict__ x, const float* __restrict__ maskX,
    const float* __restrict__ b_ih, const float* __restrict__ b_hh,
    const float* __restrict__ W_out, const float* __restrict__ b_out,
    const unsigned short* __restrict__ wpack, float* __restrict__ out) {
  // h as bf16, padded row stride 264 (528B = 33*16B) -> conflict-free b128 loads
  __shared__ __align__(16) unsigned short hA[16][264];
  __shared__ __align__(16) float xS[16][36];
  __shared__ __align__(16) float maskS[16][36];
  __shared__ float WoutS[NHH];
  __shared__ float outPart[16][16];

  const int tid  = threadIdx.x;
  const int wave = tid >> 5;
  const int lane = tid & 31;
  const int bbase = blockIdx.x * 16;

  // ---- one-time init ----
  for (int i = tid; i < 16 * 264; i += 256) ((unsigned short*)hA)[i] = 0;
  if (tid < NHH) WoutS[tid] = W_out[tid];
  {
    int row = tid >> 4, c0 = (tid & 15) * 2;
    int b = bbase + row; if (b > NGRID - 1) b = NGRID - 1;
    const float2 mv = *(const float2*)(maskX + (size_t)b * NXX + c0);
    maskS[row][c0] = mv.x; maskS[row][c0 + 1] = mv.y;
  }
  float biasv[4][2];
#pragma unroll
  for (int q = 0; q < 4; ++q)
#pragma unroll
    for (int j = 0; j < 2; ++j) {
      int col = q * NHH + wave * 32 + j * 16 + (lane & 15);
      biasv[q][j] = b_ih[col] + b_hh[col];
    }
  const float bo = b_out[0];

  v8f cfrag[2];
#pragma unroll
  for (int j = 0; j < 2; ++j)
#pragma unroll
    for (int e = 0; e < 8; ++e) cfrag[j][e] = 0.0f;

  const int row16 = lane & 15;
  const int khalf = (lane < 16) ? 0 : 8;     // A-layout K offset per lane half
  const int rb    = (lane < 16) ? 0 : 8;     // C/D-layout M offset per lane half
  const GAS unsigned short* wbase = (const GAS unsigned short*)wpack;

  __syncthreads();

  for (int t = 0; t < NTT; ++t) {
    // Launder the weight base pointer (keeping addrspace(1)!) so the fragment
    // loads below are NOT loop-invariant: blocks LICM from hoisting 72
    // fragments (576 VGPRs) out of the time loop, while still selecting
    // global_load_b128 (saddr + imm offset) instead of flat_load.
    const GAS unsigned short* wp = wbase;
    asm volatile("" : "+s"(wp));
    const GAS unsigned short* wi = wp + 64 * 8 * 512;

    // stage masked x tile (16x32 fp32) into LDS
    {
      int row = tid >> 4, c0 = (tid & 15) * 2;
      int b = bbase + row; if (b > NGRID - 1) b = NGRID - 1;
      const float2 xv = *(const float2*)(x + ((size_t)t * NGRID + b) * NXX + c0);
      xS[row][c0]     = xv.x * maskS[row][c0];
      xS[row][c0 + 1] = xv.y * maskS[row][c0 + 1];
    }
    __syncthreads();   // B1: xS ready

    // A fragment for x projection (16x32 bf16)
    v16bf ax;
    {
      U16x16 u;
#pragma unroll
      for (int e = 0; e < 8; ++e) u.v[e]     = f2bfu(xS[row16][khalf + e]);
#pragma unroll
      for (int e = 0; e < 8; ++e) u.v[8 + e] = f2bfu(xS[row16][khalf + 16 + e]);
      ax = __builtin_bit_cast(v16bf, u);
    }

    // accumulators = bias broadcast (per-lane scalar, depends only on N)
    v8f acc[4][2];
#pragma unroll
    for (int q = 0; q < 4; ++q)
#pragma unroll
      for (int j = 0; j < 2; ++j) {
        float bv = biasv[q][j];
#pragma unroll
        for (int e = 0; e < 8; ++e) acc[q][j][e] = bv;
      }

    // x projection: 1 WMMA per column tile
#pragma unroll
    for (int q = 0; q < 4; ++q)
#pragma unroll
      for (int j = 0; j < 2; ++j) {
        int tile = q * 16 + wave * 2 + j;
        v16bf bf = ld_frag_g(wi + (size_t)tile * 512 + lane * 16);
        acc[q][j] = __builtin_amdgcn_wmma_f32_16x16x32_bf16(
            false, ax, false, bf, (short)0, acc[q][j], false, false);
      }

    // recurrent GEMM: K=256 in 8 chunks of 32
    for (int kc = 0; kc < 8; ++kc) {
      v16bf ah = ld_frag2(&hA[row16][kc * 32 + khalf],
                          &hA[row16][kc * 32 + khalf + 16]);
#pragma unroll
      for (int q = 0; q < 4; ++q)
#pragma unroll
        for (int j = 0; j < 2; ++j) {
          int tile = q * 16 + wave * 2 + j;
          v16bf bf = ld_frag_g(wp + (size_t)(tile * 8 + kc) * 512 + lane * 16);
          acc[q][j] = __builtin_amdgcn_wmma_f32_16x16x32_bf16(
              false, ah, false, bf, (short)0, acc[q][j], false, false);
        }
    }
    __syncthreads();   // B2: all waves done reading old h from LDS

    // elementwise gate update + write new h (bf16) back to LDS
#pragma unroll
    for (int j = 0; j < 2; ++j) {
      v8f vi, vf, vg, vo, hcur;
#pragma unroll
      for (int e = 0; e < 8; ++e) {
        vi[e] = sigf(acc[0][j][e]);
        vf[e] = sigf(acc[1][j][e]);
        vg[e] = tanhf(acc[2][j][e]);
        vo[e] = sigf(acc[3][j][e]);
      }
      v8f cn = vf * cfrag[j] + vi * vg;
      cfrag[j] = cn;
#pragma unroll
      for (int e = 0; e < 8; ++e) hcur[e] = vo[e] * tanhf(cn[e]);
      int colh = wave * 32 + j * 16 + (lane & 15);
#pragma unroll
      for (int r = 0; r < 8; ++r) hA[rb + r][colh] = f2bfu(hcur[r]);
    }
    __syncthreads();   // B3: new h visible

    // fused output projection: out[t,b] = h . W_out + b_out
    {
      int row = tid >> 4, seg = tid & 15;
      float s = 0.0f;
#pragma unroll
      for (int k = 0; k < 16; ++k)
        s += bf2f(hA[row][seg * 16 + k]) * WoutS[seg * 16 + k];
      outPart[row][seg] = s;
    }
    __syncthreads();   // B4: partials visible
    if (tid < 16) {
      float s = bo;
#pragma unroll
      for (int k = 0; k < 16; ++k) s += outPart[tid][k];
      int b = bbase + tid;
      if (b < NGRID) out[(size_t)t * NGRID + b] = s;
    }
  }
}

extern "C" void kernel_launch(void* const* d_in, const int* in_sizes, int n_in,
                              void* d_out, int out_size, void* d_ws, size_t ws_size,
                              hipStream_t stream) {
  (void)in_sizes; (void)n_in; (void)out_size; (void)ws_size;
  const float* x     = (const float*)d_in[0];
  const float* maskX = (const float*)d_in[1];
  const float* W_ih  = (const float*)d_in[2];
  const float* W_hh  = (const float*)d_in[3];
  const float* b_ih  = (const float*)d_in[4];
  const float* b_hh  = (const float*)d_in[5];
  const float* W_out = (const float*)d_in[6];
  const float* b_out = (const float*)d_in[7];
  unsigned short* wpack = (unsigned short*)d_ws;   // needs 576 KiB scratch

  // 262144 + 32768 bf16 elements, one per thread
  prepack_weights<<<1152, 256, 0, stream>>>(W_ih, W_hh, wpack);
  // 63 batch tiles of 16 rows cover NGRID=1000
  lstm_persist<<<63, 256, 0, stream>>>(x, maskX, b_ih, b_hh, W_out, b_out,
                                       wpack, (float*)d_out);
}
